// SelectiveSSM_87703232184785
// MI455X (gfx1250) — matrix operand order repaired and checked by reading, not compiled, not run on hardware
//
#include <hip/hip_runtime.h>
#include <hip/hip_bf16.h>

// ---------------- problem constants (from reference) ----------------
#define D_MODEL   768
#define STATE_DIM 16
#define D_INNER   1536      // D_MODEL * 2
#define BATCH     2
#define SEQ_LEN   2048
#define M_TOTAL   (BATCH * SEQ_LEN)   // 4096 rows
#define NCH       16                  // scan chunks per sequence
#define CHUNK     (SEQ_LEN / NCH)     // 128 steps per chunk

typedef __attribute__((ext_vector_type(16))) __bf16 v16bf;
typedef __attribute__((ext_vector_type(8)))  __bf16 v8bf;
typedef __attribute__((ext_vector_type(8)))  float  v8f;

union bffrag { v16bf v; v8bf h[2]; };

// Epilogue modes for the blocked GEMM
#define EPI_NONE       0
#define EPI_SPLIT_SILU 1   // cols < D_INNER: store f32 + bf16 copy; cols >= D_INNER: silu in place
#define EPI_SOFTPLUS   2   // softplus(v + bias[col])

// ---------------- fp32 -> bf16 conversion (same layout) ----------------
__global__ void cvt_bf16_kernel(const float* __restrict__ in,
                                __bf16* __restrict__ out, int n) {
  int i = blockIdx.x * blockDim.x + threadIdx.x;
  if (i < n) out[i] = (__bf16)in[i];
}

// ---------------- fp32 [K,N] -> bf16 [N,K] (transpose) ----------------
__global__ void cvt_tr_bf16_kernel(const float* __restrict__ in,
                                   __bf16* __restrict__ out, int K, int N) {
  int i = blockIdx.x * blockDim.x + threadIdx.x;
  if (i < K * N) {
    int k = i / N, n = i % N;
    out[(size_t)n * K + k] = (__bf16)in[i];
  }
}

// ---------------- fragment epilogue store (uniform branch per 16-col tile) ----------------
template <int EPI>
__device__ __forceinline__ void store_tile(float* __restrict__ C, const v8f& acc,
                                           int rowBase, int colBase, int mn, int half,
                                           int ldc, __bf16* __restrict__ aux_bf,
                                           const float* __restrict__ bias) {
  const int col = colBase + mn;
#pragma unroll
  for (int r = 0; r < 8; ++r) {
    const int row = rowBase + half * 8 + r;
    float v = acc[r];
    if (EPI == EPI_SPLIT_SILU) {
      if (colBase < D_INNER) {                       // x_p half (uniform per tile)
        C[(size_t)row * ldc + col] = v;
        aux_bf[(size_t)row * D_INNER + col] = (__bf16)v;
      } else {                                       // z half -> silu(z)
        C[(size_t)row * ldc + col] = v / (1.0f + __expf(-v));
      }
    } else if (EPI == EPI_SOFTPLUS) {
      float t = v + bias[col];
      C[(size_t)row * ldc + col] = (t > 20.0f) ? t : log1pf(__expf(t));
    } else {
      C[(size_t)row * ldc + col] = v;
    }
  }
}

// ---------------- 2x2 register-blocked WMMA bf16 GEMM ----------------
// C[M,N] = A[M,K] * Bt[N,K]^T.  256 threads = 8 waves, waves 2(M) x 4(N),
// each wave owns a 32x32 tile (4 x v8f accumulators) -> block tile 64 x 128.
template <int EPI>
__global__ __launch_bounds__(256) void gemm32_bf16_wmma_kernel(
    const __bf16* __restrict__ A, const __bf16* __restrict__ Bt,
    float* __restrict__ C, int K, int ldc,
    __bf16* __restrict__ aux_bf, const float* __restrict__ bias) {
  const int lane = threadIdx.x & 31;
  const int wave = threadIdx.x >> 5;
  const int tileM = blockIdx.y * 64  + (wave >> 2) * 32;
  const int tileN = blockIdx.x * 128 + (wave & 3) * 32;
  const int mn   = lane & 15;
  const int half = lane >> 4;

  const __bf16* arow0 = A  + (size_t)(tileM + mn) * K + half * 8;   // rows tileM..+15
  const __bf16* arow1 = arow0 + (size_t)16 * K;                     // rows +16..+31
  const __bf16* brow0 = Bt + (size_t)(tileN + mn) * K + half * 8;   // cols tileN..+15
  const __bf16* brow1 = brow0 + (size_t)16 * K;                     // cols +16..+31

  v8f acc00 = {}, acc01 = {}, acc10 = {}, acc11 = {};
  for (int k0 = 0; k0 < K; k0 += 32) {
    bffrag a0, a1, b0, b1;
    a0.h[0] = *(const v8bf*)(arow0 + k0);
    a0.h[1] = *(const v8bf*)(arow0 + k0 + 16);
    a1.h[0] = *(const v8bf*)(arow1 + k0);
    a1.h[1] = *(const v8bf*)(arow1 + k0 + 16);
    b0.h[0] = *(const v8bf*)(brow0 + k0);
    b0.h[1] = *(const v8bf*)(brow0 + k0 + 16);
    b1.h[0] = *(const v8bf*)(brow1 + k0);
    b1.h[1] = *(const v8bf*)(brow1 + k0 + 16);
    __builtin_prefetch(arow0 + k0 + 256, 0, 1);   // global_prefetch_b8
    __builtin_prefetch(arow1 + k0 + 256, 0, 1);
    __builtin_prefetch(brow0 + k0 + 256, 0, 1);
    __builtin_prefetch(brow1 + k0 + 256, 0, 1);
    acc00 = __builtin_amdgcn_wmma_f32_16x16x32_bf16(false, a0.v, false, b0.v, (short)0, acc00, false, false);
    acc01 = __builtin_amdgcn_wmma_f32_16x16x32_bf16(false, a0.v, false, b1.v, (short)0, acc01, false, false);
    acc10 = __builtin_amdgcn_wmma_f32_16x16x32_bf16(false, a1.v, false, b0.v, (short)0, acc10, false, false);
    acc11 = __builtin_amdgcn_wmma_f32_16x16x32_bf16(false, a1.v, false, b1.v, (short)0, acc11, false, false);
  }
  store_tile<EPI>(C, acc00, tileM,      tileN,      mn, half, ldc, aux_bf, bias);
  store_tile<EPI>(C, acc01, tileM,      tileN + 16, mn, half, ldc, aux_bf, bias);
  store_tile<EPI>(C, acc10, tileM + 16, tileN,      mn, half, ldc, aux_bf, bias);
  store_tile<EPI>(C, acc11, tileM + 16, tileN + 16, mn, half, ldc, aux_bf, bias);
}

// ---------------- single-tile WMMA GEMM (for N = 32, the B/C projection) ----------------
__global__ __launch_bounds__(256) void gemm16_bf16_wmma_kernel(
    const __bf16* __restrict__ A, const __bf16* __restrict__ Bt,
    float* __restrict__ C, int K, int ldc) {
  const int lane = threadIdx.x & 31;
  const int wave = threadIdx.x >> 5;
  const int tileM = blockIdx.y * 64 + (wave >> 1) * 16;
  const int tileN = blockIdx.x * 32 + (wave & 1) * 16;
  const int mn   = lane & 15;
  const int half = lane >> 4;

  const __bf16* arow = A  + (size_t)(tileM + mn) * K + half * 8;
  const __bf16* brow = Bt + (size_t)(tileN + mn) * K + half * 8;

  v8f acc = {};
  for (int k0 = 0; k0 < K; k0 += 32) {
    bffrag a, b;
    a.h[0] = *(const v8bf*)(arow + k0);
    a.h[1] = *(const v8bf*)(arow + k0 + 16);
    b.h[0] = *(const v8bf*)(brow + k0);
    b.h[1] = *(const v8bf*)(brow + k0 + 16);
    __builtin_prefetch(arow + k0 + 256, 0, 1);
    __builtin_prefetch(brow + k0 + 256, 0, 1);
    acc = __builtin_amdgcn_wmma_f32_16x16x32_bf16(false, a.v, false, b.v, (short)0, acc, false, false);
  }
#pragma unroll
  for (int r = 0; r < 8; ++r)
    C[(size_t)(tileM + half * 8 + r) * ldc + tileN + mn] = acc[r];
}

// ================= chunked selective scan (3 phases) =================
// Recurrence: h_t = exp(dt_t*A)*h_{t-1} + (x_t*dt_t)*B_t. Linear in h, and the
// per-chunk decay product telescopes: prod_t exp(dt_t*A[n]) = exp(A[n]*sum dt).
// Phase 1: per (b, chunk, d) local scan from h=0 -> h_end[16], sum(dt).
// Phase 2: per (b, d) combine 16 chunk aggregates -> carry-in per chunk.
// Phase 3: per (b, chunk, d) local scan seeded with carry-in -> gated output.
// Aggregate layout [b, c, n, d] (d fastest) for coalesced lane access.

// ---- phase 1: chunk-local scans ----
__global__ __launch_bounds__(256) void scan_phase1_kernel(
    const float* __restrict__ xz,   // [M_TOTAL, 2*D_INNER]: x_p | silu(z)
    const float* __restrict__ dt,   // [M_TOTAL, D_INNER]
    const float* __restrict__ BC,   // [M_TOTAL, 2*STATE_DIM]
    const float* __restrict__ A_log,
    float* __restrict__ hend,       // [BATCH, NCH, STATE_DIM, D_INNER]
    float* __restrict__ dtsum) {    // [BATCH, NCH, D_INNER]
  int tid = blockIdx.x * blockDim.x + threadIdx.x;   // over BATCH*NCH*D_INNER
  if (tid >= BATCH * NCH * D_INNER) return;
  int d = tid % D_INNER;
  int c = (tid / D_INNER) % NCH;
  int b = tid / (D_INNER * NCH);

  float A[STATE_DIM], h[STATE_DIM];
#pragma unroll
  for (int n = 0; n < STATE_DIM; ++n) { A[n] = -__expf(A_log[n]); h[n] = 0.0f; }

  float s = 0.0f;
  const int t0 = c * CHUNK;
  for (int t = t0; t < t0 + CHUNK; ++t) {
    size_t row = (size_t)b * SEQ_LEN + t;
    float dtv = dt[row * D_INNER + d];
    float xv  = xz[row * (2 * D_INNER) + d];
    const float* bc = BC + row * (2 * STATE_DIM);
    float xdt = xv * dtv;
    s += dtv;
#pragma unroll
    for (int n = 0; n < STATE_DIM; ++n)
      h[n] = fmaf(__expf(dtv * A[n]), h[n], xdt * bc[n]);
  }
  size_t base = ((size_t)(b * NCH + c) * STATE_DIM) * D_INNER + d;
#pragma unroll
  for (int n = 0; n < STATE_DIM; ++n) hend[base + (size_t)n * D_INNER] = h[n];
  dtsum[(size_t)(b * NCH + c) * D_INNER + d] = s;
}

// ---- phase 2: combine chunk aggregates -> carry-in per chunk ----
__global__ __launch_bounds__(256) void scan_phase2_kernel(
    const float* __restrict__ A_log,
    const float* __restrict__ hend,   // [BATCH, NCH, STATE_DIM, D_INNER]
    const float* __restrict__ dtsum,  // [BATCH, NCH, D_INNER]
    float* __restrict__ hcarry) {     // [BATCH, NCH, STATE_DIM, D_INNER] (carry-in)
  int tid = blockIdx.x * blockDim.x + threadIdx.x;   // over BATCH*D_INNER
  if (tid >= BATCH * D_INNER) return;
  int b = tid / D_INNER;
  int d = tid % D_INNER;

  float A[STATE_DIM], carry[STATE_DIM];
#pragma unroll
  for (int n = 0; n < STATE_DIM; ++n) { A[n] = -__expf(A_log[n]); carry[n] = 0.0f; }

  for (int c = 0; c < NCH; ++c) {
    size_t base = ((size_t)(b * NCH + c) * STATE_DIM) * D_INNER + d;
#pragma unroll
    for (int n = 0; n < STATE_DIM; ++n) hcarry[base + (size_t)n * D_INNER] = carry[n];
    float s = dtsum[(size_t)(b * NCH + c) * D_INNER + d];
#pragma unroll
    for (int n = 0; n < STATE_DIM; ++n)
      carry[n] = fmaf(__expf(A[n] * s), carry[n], hend[base + (size_t)n * D_INNER]);
  }
}

// ---- phase 3: seeded local scans + readout + gate ----
__global__ __launch_bounds__(256) void scan_phase3_kernel(
    const float* __restrict__ xz,
    const float* __restrict__ dt,
    const float* __restrict__ BC,
    const float* __restrict__ A_log,
    const float* __restrict__ Dvec,
    const float* __restrict__ hcarry, // [BATCH, NCH, STATE_DIM, D_INNER]
    __bf16* __restrict__ ybf) {       // [M_TOTAL, D_INNER]
  int tid = blockIdx.x * blockDim.x + threadIdx.x;   // over BATCH*NCH*D_INNER
  if (tid >= BATCH * NCH * D_INNER) return;
  int d = tid % D_INNER;
  int c = (tid / D_INNER) % NCH;
  int b = tid / (D_INNER * NCH);

  float A[STATE_DIM], h[STATE_DIM];
  size_t base = ((size_t)(b * NCH + c) * STATE_DIM) * D_INNER + d;
#pragma unroll
  for (int n = 0; n < STATE_DIM; ++n) {
    A[n] = -__expf(A_log[n]);
    h[n] = hcarry[base + (size_t)n * D_INNER];
  }
  float Dd = Dvec[d];

  const int t0 = c * CHUNK;
  for (int t = t0; t < t0 + CHUNK; ++t) {
    size_t row = (size_t)b * SEQ_LEN + t;
    float dtv = dt[row * D_INNER + d];
    float xv  = xz[row * (2 * D_INNER) + d];
    float sz  = xz[row * (2 * D_INNER) + D_INNER + d];
    const float* bc = BC + row * (2 * STATE_DIM);
    float xdt = xv * dtv;
    float acc = 0.0f;
#pragma unroll
    for (int n = 0; n < STATE_DIM; ++n) {
      h[n] = fmaf(__expf(dtv * A[n]), h[n], xdt * bc[n]);
      acc  = fmaf(h[n], bc[STATE_DIM + n], acc);
    }
    ybf[row * D_INNER + d] = (__bf16)((acc + xv * Dd) * sz);
  }
}

// ---------------- host-side launch ----------------
extern "C" void kernel_launch(void* const* d_in, const int* in_sizes, int n_in,
                              void* d_out, int out_size, void* d_ws, size_t ws_size,
                              hipStream_t stream) {
  (void)in_sizes; (void)n_in; (void)out_size; (void)ws_size;
  const float* x     = (const float*)d_in[0];
  const float* W_in  = (const float*)d_in[1];
  const float* W_x   = (const float*)d_in[2];
  const float* W_dt  = (const float*)d_in[3];
  const float* b_dt  = (const float*)d_in[4];
  const float* A_log = (const float*)d_in[5];
  const float* Dvec  = (const float*)d_in[6];
  const float* W_out = (const float*)d_in[7];
  float* out = (float*)d_out;

  char* ws = (char*)d_ws;
  size_t off = 0;
  auto take = [&](size_t bytes) { char* p = ws + off; off += (bytes + 255) & ~(size_t)255; return p; };

  float*  xz     = (float*)take((size_t)M_TOTAL * 2 * D_INNER * 4); // x_p | silu(z)
  float*  dt     = (float*)take((size_t)M_TOTAL * D_INNER * 4);
  float*  BC     = (float*)take((size_t)M_TOTAL * 2 * STATE_DIM * 4);
  float*  hend   = (float*)take((size_t)BATCH * NCH * STATE_DIM * D_INNER * 4);
  float*  hcarry = (float*)take((size_t)BATCH * NCH * STATE_DIM * D_INNER * 4);
  float*  dtsum  = (float*)take((size_t)BATCH * NCH * D_INNER * 4);
  __bf16* x_bf   = (__bf16*)take((size_t)M_TOTAL * D_MODEL * 2);
  __bf16* xp_bf  = (__bf16*)take((size_t)M_TOTAL * D_INNER * 2);
  __bf16* y_bf   = (__bf16*)take((size_t)M_TOTAL * D_INNER * 2);
  __bf16* WinT   = (__bf16*)take((size_t)(2 * D_INNER) * D_MODEL * 2);   // [3072, 768]
  __bf16* WxT    = (__bf16*)take((size_t)(2 * STATE_DIM) * D_INNER * 2); // [32, 1536]
  __bf16* WdtT   = (__bf16*)take((size_t)D_INNER * D_INNER * 2);         // [1536, 1536]
  __bf16* WoutT  = (__bf16*)take((size_t)D_MODEL * D_INNER * 2);         // [768, 1536]

  const int T = 256;
  // 1) conversions / transposes
  {
    int n = M_TOTAL * D_MODEL;
    cvt_bf16_kernel<<<(n + T - 1) / T, T, 0, stream>>>(x, x_bf, n);
  }
  cvt_tr_bf16_kernel<<<(D_MODEL * 2 * D_INNER + T - 1) / T, T, 0, stream>>>(W_in, WinT, D_MODEL, 2 * D_INNER);
  cvt_tr_bf16_kernel<<<(D_INNER * 2 * STATE_DIM + T - 1) / T, T, 0, stream>>>(W_x, WxT, D_INNER, 2 * STATE_DIM);
  cvt_tr_bf16_kernel<<<(D_INNER * D_INNER + T - 1) / T, T, 0, stream>>>(W_dt, WdtT, D_INNER, D_INNER);
  cvt_tr_bf16_kernel<<<(D_INNER * D_MODEL + T - 1) / T, T, 0, stream>>>(W_out, WoutT, D_INNER, D_MODEL);

  // 2) xz = x @ W_in : [4096, 3072], fused split + bf16(x_p) + silu(z)
  gemm32_bf16_wmma_kernel<EPI_SPLIT_SILU>
      <<<dim3(2 * D_INNER / 128, M_TOTAL / 64), T, 0, stream>>>(
          x_bf, WinT, xz, /*K=*/D_MODEL, /*ldc=*/2 * D_INNER, xp_bf, nullptr);

  // 3) dt = softplus(x_p @ W_dt + b_dt) : [4096, 1536], fused epilogue
  gemm32_bf16_wmma_kernel<EPI_SOFTPLUS>
      <<<dim3(D_INNER / 128, M_TOTAL / 64), T, 0, stream>>>(
          xp_bf, WdtT, dt, /*K=*/D_INNER, /*ldc=*/D_INNER, nullptr, b_dt);

  // 4) BC = x_p @ W_x : [4096, 32]
  gemm16_bf16_wmma_kernel<<<dim3(1, M_TOTAL / 64), T, 0, stream>>>(
      xp_bf, WxT, BC, /*K=*/D_INNER, /*ldc=*/2 * STATE_DIM);

  // 5) chunked selective scan -> gated y (bf16)
  {
    int n1 = BATCH * NCH * D_INNER;   // 49152 threads, 1536 waves
    scan_phase1_kernel<<<(n1 + T - 1) / T, T, 0, stream>>>(xz, dt, BC, A_log, hend, dtsum);
    int n2 = BATCH * D_INNER;         // 3072 threads, tiny combine
    scan_phase2_kernel<<<(n2 + T - 1) / T, T, 0, stream>>>(A_log, hend, dtsum, hcarry);
    scan_phase3_kernel<<<(n1 + T - 1) / T, T, 0, stream>>>(xz, dt, BC, A_log, Dvec, hcarry, y_bf);
  }

  // 6) out = y @ W_out : [4096, 768]
  gemm32_bf16_wmma_kernel<EPI_NONE>
      <<<dim3(D_MODEL / 128, M_TOTAL / 64), T, 0, stream>>>(
          y_bf, WoutT, out, /*K=*/D_INNER, /*ldc=*/D_MODEL, nullptr, nullptr);
}
